// MultiheadAttention_76897094467801
// MI455X (gfx1250) — matrix operand order repaired and checked by reading, not compiled, not run on hardware
//
#include <hip/hip_runtime.h>

typedef unsigned short u16;
typedef unsigned int   u32;
typedef unsigned long long u64;
typedef __attribute__((ext_vector_type(16))) __bf16 v16bf;
typedef __attribute__((ext_vector_type(8)))  float  v8f;

#define B_   2
#define S_   2048
#define D_   1024
#define H_   16
#define HD_  64
#define N3_  3072   // 3*D

union FragBF {
  v16bf v;
  uint4 q[2];
  u16   u[16];
};

__device__ __forceinline__ u16 f2bf(float f) {
  u32 u = __float_as_uint(f);
  u32 r = u + 0x7FFFu + ((u >> 16) & 1u);   // round-to-nearest-even
  return (u16)(r >> 16);
}

// Async DMA: global -> LDS, 16 bytes per lane, tracked by ASYNCcnt.
__device__ __forceinline__ void async_b128(const u16* lds_dst, const u16* gsrc) {
  u32 lds_off = (u32)(uintptr_t)lds_dst;          // low 32 bits = LDS offset
  u64 gaddr   = (u64)(uintptr_t)gsrc;
  asm volatile("global_load_async_to_lds_b128 %0, %1, off"
               :: "v"(lds_off), "v"(gaddr) : "memory");
}

// ---------------------------------------------------------------- convert
__global__ void cvt_f32_bf16(const float* __restrict__ in, u16* __restrict__ out, int n) {
  int i = blockIdx.x * blockDim.x + threadIdx.x;
  if (i < n) out[i] = f2bf(in[i]);
}

// ---------------------------------------------------------------- wave GEMM core
// One wave computes a 64x64 tile of C = A[M,K] * B[N,K]^T  (both row-major, K contiguous)
__device__ __forceinline__ void wave_gemm_64x64(const u16* __restrict__ A,
                                                const u16* __restrict__ Bm,
                                                int K, int mbase, int nbase,
                                                int lane, v8f acc[4][4]) {
  const int ln = lane & 15;
  const int kh = lane >> 4;
  for (int k0 = 0; k0 < K; k0 += 32) {
    FragBF a[4];
#pragma unroll
    for (int mi = 0; mi < 4; ++mi) {
      // A-frag: lane = M row; elems 0..7 = K[k0+kh*8 ..], elems 8..15 = K[k0+16+kh*8 ..]
      const u16* ap = A + (size_t)(mbase + mi * 16 + ln) * K + k0 + kh * 8;
      a[mi].q[0] = *(const uint4*)ap;
      a[mi].q[1] = *(const uint4*)(ap + 16);
    }
#pragma unroll
    for (int ni = 0; ni < 4; ++ni) {
      // B-frag: lane = N col; elems 0..7 = K[k0+kh*8 ..], elems 8..15 = K[k0+16+kh*8 ..]
      FragBF b;
      const u16* bp = Bm + (size_t)(nbase + ni * 16 + ln) * K + k0 + kh * 8;
      b.q[0] = *(const uint4*)bp;
      b.q[1] = *(const uint4*)(bp + 16);
#pragma unroll
      for (int mi = 0; mi < 4; ++mi) {
        acc[mi][ni] = __builtin_amdgcn_wmma_f32_16x16x32_bf16(
            false, a[mi].v, false, b.v, (short)0, acc[mi][ni], false, false);
      }
    }
  }
}

// ---------------------------------------------------------------- QKV projection
// qkv = x @ w_qkv^T + b_qkv, scattered into:
//   q_s[b,h,s,d] (pre-scaled by 1/sqrt(hd)), k_s[b,h,s,d], v_t[b,h,d,s] (transposed!)
__global__ void __launch_bounds__(256) qkv_gemm(const u16* __restrict__ x,
                                                const u16* __restrict__ wqkv,
                                                const float* __restrict__ bias,
                                                u16* __restrict__ qs,
                                                u16* __restrict__ ks,
                                                u16* __restrict__ vt) {
  const int lane = threadIdx.x & 31;
  const int wid  = threadIdx.x >> 5;
  const int mbase = blockIdx.y * 128 + (wid >> 2) * 64;
  const int nbase = blockIdx.x * 256 + (wid & 3) * 64;

  v8f acc[4][4];
#pragma unroll
  for (int mi = 0; mi < 4; ++mi)
#pragma unroll
    for (int ni = 0; ni < 4; ++ni)
#pragma unroll
      for (int e = 0; e < 8; ++e) acc[mi][ni][e] = 0.f;

  wave_gemm_64x64(x, wqkv, D_, mbase, nbase, lane, acc);

  const int ln = lane & 15, half = lane >> 4;
  const int b = mbase >> 11;  // 2048 tokens per batch, block never straddles
#pragma unroll
  for (int ni = 0; ni < 4; ++ni) {
    const int j = nbase + ni * 16 + ln;   // output column, fixed per lane
    const float bj = bias[j];
    const int h = j / 192;
    const int c = j - h * 192;            // 0..63 q, 64..127 k, 128..191 v
#pragma unroll
    for (int mi = 0; mi < 4; ++mi) {
#pragma unroll
      for (int r = 0; r < 8; ++r) {
        const int t = mbase + mi * 16 + r + 8 * half;
        const int s = t & (S_ - 1);
        const float v = acc[mi][ni][r] + bj;
        if (c < 64) {
          qs[(size_t)((b * H_ + h) * S_ + s) * HD_ + c] = f2bf(v * 0.125f);
        } else if (c < 128) {
          ks[(size_t)((b * H_ + h) * S_ + s) * HD_ + (c - 64)] = f2bf(v);
        } else {
          vt[(size_t)((b * H_ + h) * HD_ + (c - 128)) * S_ + s] = f2bf(v);
        }
      }
    }
  }
}

// ---------------------------------------------------------------- flash attention
// One block = 4 waves = 64 Q rows for one (b,h). Online softmax over key chunks of 32.
// K/V chunks staged once per block into LDS via async DMA, double-buffered.
__global__ void __launch_bounds__(128) attn_kernel(const u16* __restrict__ qs,
                                                   const u16* __restrict__ ks,
                                                   const u16* __restrict__ vt,
                                                   u16* __restrict__ attn) {
  __shared__ __align__(16) u16 kbuf[2][32][64];   // keys x d      (2 x 4KB)
  __shared__ __align__(16) u16 vbuf[2][64][32];   // d x keys      (2 x 4KB)
  __shared__ __align__(16) u16 plds[4][16][32];   // per-wave P bounce (4KB)

  const int tid  = threadIdx.x;
  const int lane = tid & 31;
  const int wid  = tid >> 5;
  const int ln = lane & 15, kh = lane >> 4;
  const int h = blockIdx.y, b = blockIdx.z;
  const int qbase = blockIdx.x * 64 + wid * 16;

  const u16* qp = qs + (size_t)((b * H_ + h) * S_) * HD_;
  const u16* kp = ks + (size_t)((b * H_ + h) * S_) * HD_;
  const u16* vp = vt + (size_t)((b * H_ + h) * HD_) * S_;

  // Q fragments (hd=64 -> two K=32 slabs), loaded once
  FragBF qf[2];
#pragma unroll
  for (int j = 0; j < 2; ++j) {
    const u16* ap = qp + (size_t)(qbase + ln) * HD_ + j * 32 + kh * 8;
    qf[j].q[0] = *(const uint4*)ap;
    qf[j].q[1] = *(const uint4*)(ap + 16);
  }

  v8f acc[4];
#pragma unroll
  for (int n = 0; n < 4; ++n)
#pragma unroll
    for (int e = 0; e < 8; ++e) acc[n][e] = 0.f;

  float mrow[8], lrow[8];
#pragma unroll
  for (int r = 0; r < 8; ++r) { mrow[r] = -1e30f; lrow[r] = 0.f; }

  // issue async K/V DMA for a chunk: 4 x b128 per thread = 8KB per block
  auto issue_chunk = [&](int kb, int buf) {
#pragma unroll
    for (int j = 0; j < 2; ++j) {
      const int flat = j * 1024 + tid * 8;              // half-word index
      const int krow = flat >> 6, kcol = flat & 63;     // K tile 32x64
      async_b128(&kbuf[buf][krow][kcol], kp + (size_t)(kb + krow) * HD_ + kcol);
      const int dro = flat >> 5, kj = flat & 31;        // V tile 64x32
      async_b128(&vbuf[buf][dro][kj], vp + (size_t)dro * S_ + kb + kj);
    }
  };

  issue_chunk(0, 0);

  int cur = 0;
  for (int kb = 0; kb < S_; kb += 32, cur ^= 1) {
    if (kb + 32 < S_) {
      issue_chunk(kb + 32, cur ^ 1);                    // prefetch next chunk
      asm volatile("s_wait_asynccnt 4" ::: "memory");   // current chunk landed
    } else {
      asm volatile("s_wait_asynccnt 0" ::: "memory");
    }
    __syncthreads();                                     // LDS visible block-wide

    // logits: two 16x16 tiles (keys kb..kb+15 and kb+16..kb+31), K-frags from LDS
    v8f s[2];
#pragma unroll
    for (int t = 0; t < 2; ++t) {
#pragma unroll
      for (int e = 0; e < 8; ++e) s[t][e] = 0.f;
#pragma unroll
      for (int j = 0; j < 2; ++j) {
        FragBF kf;
        const u16* bp = &kbuf[cur][t * 16 + ln][j * 32 + kh * 8];
        kf.q[0] = *(const uint4*)bp;
        kf.q[1] = *(const uint4*)(bp + 16);
        s[t] = __builtin_amdgcn_wmma_f32_16x16x32_bf16(
            false, qf[j].v, false, kf.v, (short)0, s[t], false, false);
      }
    }

    // online softmax: row r lives in lanes 0-15 (rows 0..7) / 16-31 (rows 8..15)
#pragma unroll
    for (int r = 0; r < 8; ++r) {
      float cm = fmaxf(s[0][r], s[1][r]);
#pragma unroll
      for (int off = 8; off > 0; off >>= 1) cm = fmaxf(cm, __shfl_xor(cm, off, 32));
      const float mnew  = fmaxf(mrow[r], cm);
      const float alpha = __expf(mrow[r] - mnew);
      const float p0 = __expf(s[0][r] - mnew);
      const float p1 = __expf(s[1][r] - mnew);
      float lsum = p0 + p1;
#pragma unroll
      for (int off = 8; off > 0; off >>= 1) lsum += __shfl_xor(lsum, off, 32);
      mrow[r] = mnew;
      lrow[r] = lrow[r] * alpha + lsum;
#pragma unroll
      for (int n = 0; n < 4; ++n) acc[n][r] *= alpha;
      const int row = r + 8 * kh;
      plds[wid][row][ln]      = f2bf(p0);
      plds[wid][row][ln + 16] = f2bf(p1);
    }
    // relayout P: C-fragment -> A-fragment via per-wave LDS (DS pipe is in-order)
    FragBF pf;
    pf.q[0] = *(const uint4*)&plds[wid][ln][kh * 8];
    pf.q[1] = *(const uint4*)&plds[wid][ln][16 + kh * 8];

    // acc += P(16x32) x V(32x64); V-frags from LDS (pre-transposed layout)
#pragma unroll
    for (int n = 0; n < 4; ++n) {
      FragBF vf;
      const u16* bp = &vbuf[cur][n * 16 + ln][kh * 8];
      vf.q[0] = *(const uint4*)bp;
      vf.q[1] = *(const uint4*)(bp + 16);
      acc[n] = __builtin_amdgcn_wmma_f32_16x16x32_bf16(
          false, pf.v, false, vf.v, (short)0, acc[n], false, false);
    }
    __syncthreads();   // all readers of buf `cur` done before it is re-filled
  }

  // normalize and store attn[b, s, h*64 + d] as bf16 (A-operand for final GEMM)
#pragma unroll
  for (int n = 0; n < 4; ++n) {
#pragma unroll
    for (int r = 0; r < 8; ++r) {
      const int qrow = qbase + r + 8 * kh;
      const float o = acc[n][r] / lrow[r];
      attn[(size_t)(b * S_ + qrow) * D_ + h * HD_ + n * 16 + ln] = f2bf(o);
    }
  }
}

// ---------------------------------------------------------------- output projection
__global__ void __launch_bounds__(256) out_gemm(const u16* __restrict__ attn,
                                                const u16* __restrict__ wo,
                                                const float* __restrict__ bias,
                                                float* __restrict__ out) {
  const int lane = threadIdx.x & 31;
  const int wid  = threadIdx.x >> 5;
  const int mbase = blockIdx.y * 128 + (wid >> 2) * 64;
  const int nbase = blockIdx.x * 256 + (wid & 3) * 64;

  v8f acc[4][4];
#pragma unroll
  for (int mi = 0; mi < 4; ++mi)
#pragma unroll
    for (int ni = 0; ni < 4; ++ni)
#pragma unroll
      for (int e = 0; e < 8; ++e) acc[mi][ni][e] = 0.f;

  wave_gemm_64x64(attn, wo, D_, mbase, nbase, lane, acc);

  const int ln = lane & 15, half = lane >> 4;
#pragma unroll
  for (int ni = 0; ni < 4; ++ni) {
    const int j = nbase + ni * 16 + ln;
    const float bj = bias[j];
#pragma unroll
    for (int mi = 0; mi < 4; ++mi) {
#pragma unroll
      for (int r = 0; r < 8; ++r) {
        const int t = mbase + mi * 16 + r + 8 * half;
        out[(size_t)t * D_ + j] = acc[mi][ni][r] + bj;
      }
    }
  }
}

// ---------------------------------------------------------------- launch
extern "C" void kernel_launch(void* const* d_in, const int* in_sizes, int n_in,
                              void* d_out, int out_size, void* d_ws, size_t ws_size,
                              hipStream_t stream) {
  const float* x     = (const float*)d_in[0];   // [B,S,D]
  const float* w_qkv = (const float*)d_in[1];   // [3D,D]
  const float* b_qkv = (const float*)d_in[2];   // [3D]
  const float* w_o   = (const float*)d_in[3];   // [D,D]
  const float* b_o   = (const float*)d_in[4];   // [D]
  float* out = (float*)d_out;

  const size_t nx   = (size_t)B_ * S_ * D_;       // 4 Mi
  const size_t nwq  = (size_t)N3_ * D_;           // 3 Mi
  const size_t nwo  = (size_t)D_ * D_;            // 1 Mi
  const size_t nhsd = (size_t)B_ * H_ * S_ * HD_; // 4 Mi

  u16* ws = (u16*)d_ws;
  u16* x_bf  = ws;              ws += nx;
  u16* wq_bf = ws;              ws += nwq;
  u16* wo_bf = ws;              ws += nwo;
  u16* q_s   = ws;              ws += nhsd;
  u16* k_s   = ws;              ws += nhsd;
  u16* v_t   = ws;              ws += nhsd;
  u16* attn  = ws;              ws += nx;

  cvt_f32_bf16<<<(int)((nx  + 255) / 256), 256, 0, stream>>>(x,     x_bf,  (int)nx);
  cvt_f32_bf16<<<(int)((nwq + 255) / 256), 256, 0, stream>>>(w_qkv, wq_bf, (int)nwq);
  cvt_f32_bf16<<<(int)((nwo + 255) / 256), 256, 0, stream>>>(w_o,   wo_bf, (int)nwo);

  qkv_gemm<<<dim3(N3_ / 256, (B_ * S_) / 128), 256, 0, stream>>>(
      x_bf, wq_bf, b_qkv, q_s, k_s, v_t);

  attn_kernel<<<dim3(S_ / 64, H_, B_), 128, 0, stream>>>(q_s, k_s, v_t, attn);

  out_gemm<<<dim3(D_ / 256, (B_ * S_) / 128), 256, 0, stream>>>(
      attn, wo_bf, b_o, out);
}
